// ASGNet_33389075759546
// MI455X (gfx1250) — matrix operand order repaired and checked by reading, not compiled, not run on hardware
//
#include <hip/hip_runtime.h>
#include <hip/hip_bf16.h>
#include <math.h>

typedef __attribute__((ext_vector_type(16))) _Float16 v16h;
typedef __attribute__((ext_vector_type(8)))  float    v8f;

#define NB    8      // batch
#define NW    96     // window
#define NN    512    // series
#define NH    128    // hidden
#define HEADS 8
#define HD    16
#define BOT   6
#define SCALES 5
#define HOR   12
#define ROWS  (NB*NN)            // 4096
#define CATK  (SCALES*NH)        // 640
#define EPS_BN 1e-5f
#define EPS_LN 1e-5f

// ---------------- helpers ----------------
__device__ __forceinline__ float blockReduceSum(float v, float* red, int t) {
    red[t] = v; __syncthreads();
    for (int s = 64; s > 0; s >>= 1) { if (t < s) red[t] += red[t + s]; __syncthreads(); }
    float r = red[0]; __syncthreads();
    return r;
}
__device__ __forceinline__ unsigned f2key(float f) {
    unsigned u = __float_as_uint(f);
    return (u & 0x80000000u) ? ~u : (u | 0x80000000u);
}

// ---------------- init ----------------
__global__ void k_init(float* ent) { if (threadIdx.x == 0 && blockIdx.x == 0) ent[0] = 0.f; }

// ---------------- transpose proj_w -> f16 [128][640] ----------------
__global__ void k_prepw(const float* __restrict__ pw, _Float16* __restrict__ wT) {
    int idx = blockIdx.x * 256 + threadIdx.x;
    if (idx < CATK * NH) {
        int k = idx / NH, n = idx % NH;
        wT[n * CATK + k] = (_Float16)pw[idx];
    }
}

// ---------------- multiscale dilated conv + pool -> cat f16 [4096][640] ----------------
__global__ void k_multiscale(const float* __restrict__ x, const float* __restrict__ sw,
                             const float* __restrict__ cw, const float* __restrict__ cb,
                             const float* __restrict__ bg, const float* __restrict__ bb,
                             _Float16* __restrict__ cat) {
    __shared__ float xs[NW];
    int row = blockIdx.x; int b = row >> 9; int n = row & 511;
    int t = threadIdx.x;
    if (t < NW) xs[t] = x[((size_t)b * NW + t) * NN + n];
    __syncthreads();
    // softmax(scale_weights) redundantly per thread (5 elems)
    float swv[SCALES]; float mx = -1e30f;
    for (int i = 0; i < SCALES; i++) { swv[i] = sw[i]; mx = fmaxf(mx, swv[i]); }
    float se = 0.f;
    for (int i = 0; i < SCALES; i++) { swv[i] = expf(swv[i] - mx); se += swv[i]; }
    float inv = 1.f / se;
    int c = t;
    for (int i = 0; i < SCALES; i++) {
        int d = 1 << i;
        float w0 = cw[(i * NH + c) * 3 + 0];
        float w1 = cw[(i * NH + c) * 3 + 1];
        float w2 = cw[(i * NH + c) * 3 + 2];
        float cbv = cb[i * NH + c];
        float gsc = bg[i * NH + c] * rsqrtf(1.f + EPS_BN);
        float gb  = bb[i * NH + c];
        float acc = 0.f;
        for (int w = 0; w < NW; w++) {
            float xm = (w >= d) ? xs[w - d] : 0.f;
            float xp = (w + d < NW) ? xs[w + d] : 0.f;
            float y = fmaf(w0, xm, fmaf(w1, xs[w], w2 * xp)) + cbv;
            y = fmaf(y, gsc, gb);
            acc += fmaxf(y, 0.f);
        }
        cat[(size_t)row * CATK + i * NH + c] = (_Float16)(acc * (1.f / 96.f) * swv[i] * inv);
    }
}

// ---------------- WMMA proj GEMM [4096,640]x[640,128] + bias + BN + relu ----------------
__global__ void k_proj(const _Float16* __restrict__ cat, const _Float16* __restrict__ wT,
                       const float* __restrict__ pb, const float* __restrict__ pg,
                       const float* __restrict__ pbb, float* __restrict__ temporal) {
    int lane = threadIdx.x; int grp = lane >> 4; int mr = lane & 15;
    int rowb = blockIdx.x * 16; int colb = blockIdx.y * 16;
    const _Float16* arow = cat + (size_t)(rowb + mr) * CATK;
    const _Float16* brow = wT + (size_t)(colb + mr) * CATK;
    v8f acc = {};
    for (int k0 = 0; k0 < CATK; k0 += 32) {
        v16h a, bf;
#pragma unroll
        for (int j = 0; j < 8; j++) {
            a[j]     = arow[k0 + grp * 8 + j];
            a[j + 8] = arow[k0 + 16 + grp * 8 + j];
        }
#pragma unroll
        for (int j = 0; j < 16; j++) bf[j] = brow[k0 + ((j < 8) ? grp * 8 + j : 8 + grp * 8 + j)];
        acc = __builtin_amdgcn_wmma_f32_16x16x32_f16(false, a, false, bf, (short)0, acc, false, false);
    }
#pragma unroll
    for (int r = 0; r < 8; r++) {
        int M = r + grp * 8; int col = colb + mr;
        float v = acc[r] + pb[col];
        v = fmaf(v, pg[col] * rsqrtf(1.f + EPS_BN), pbb[col]);
        temporal[(size_t)(rowb + M) * NH + col] = fmaxf(v, 0.f);
    }
}

// ---------------- layernorm + low/high qkv -> q,k [BH,N,16] f16, vT [BH,16,N] f16 ----------------
__global__ void k_lnqkv(const float* __restrict__ temporal, const float* __restrict__ lng,
                        const float* __restrict__ lnb, const float* __restrict__ lw,
                        const float* __restrict__ lb, const float* __restrict__ hw,
                        const float* __restrict__ hb, _Float16* __restrict__ qb,
                        _Float16* __restrict__ kb, _Float16* __restrict__ vT) {
    __shared__ float xn[NH], red[NH], low[3 * BOT];
    int row = blockIdx.x; int b = row >> 9; int n = row & 511; int t = threadIdx.x;
    float v = temporal[(size_t)row * NH + t];
    float m  = blockReduceSum(v, red, t) * (1.f / 128.f);
    float vr = blockReduceSum(v * v, red, t) * (1.f / 128.f) - m * m;
    xn[t] = fmaf((v - m) * rsqrtf(vr + EPS_LN), lng[t], lnb[t]);
    __syncthreads();
    if (t < 3 * BOT) {
        float a = lb[t];
        for (int i = 0; i < NH; i++) a = fmaf(xn[i], lw[i * (3 * BOT) + t], a);
        low[t] = a;
    }
    __syncthreads();
    int h = t >> 4, dd = t & 15, bh = b * HEADS + h;
    for (int s = 0; s < 3; s++) {
        int oo = s * NH + t;
        float a = hb[oo];
        for (int j = 0; j < 3 * BOT; j++) a = fmaf(low[j], hw[j * (3 * NH) + oo], a);
        _Float16 hv = (_Float16)a;
        if (s == 0)      qb[((size_t)(bh * NN + n)) * HD + dd] = hv;
        else if (s == 1) kb[((size_t)(bh * NN + n)) * HD + dd] = hv;
        else             vT[((size_t)bh * HD + dd) * NN + n]   = hv;
    }
}

// ---------------- fused sparse attention, 4 waves/block ----------------
// wave w: score tiles ct = w*8..w*8+7; rows r = w*4..w*4+3 for select/softmax;
//         K-range [w*128, w*128+128) for attn*v with LDS partial reduction.
__global__ void __launch_bounds__(128)
k_attn(const _Float16* __restrict__ qb, const _Float16* __restrict__ kb,
       const _Float16* __restrict__ vT, const float* __restrict__ su,
       const float* __restrict__ sv, const float* __restrict__ thr,
       float* __restrict__ attb, float* __restrict__ entacc) {
    __shared__ float    sc[16][NN];      // 32 KB score strip
    __shared__ _Float16 ap[16][NN];      // 16 KB attn probs (f16)
    __shared__ float    urow[16][BOT];
    __shared__ float    pacc[4][16][16]; // 4 KB partial attn*v tiles
    int mtile = blockIdx.x, h = blockIdx.y, bz = blockIdx.z;
    int bh = bz * HEADS + h;
    int tid = threadIdx.x;
    int wid = tid >> 5;
    int lane = tid & 31, grp = lane >> 4, mr = lane & 15;

    // q A-fragment (same for all 4 waves): K in [0,16) real, [16,32) zero-padded
    v16h qa;
    {
        const _Float16* qp = qb + ((size_t)(bh * NN + mtile * 16 + mr)) * HD;
#pragma unroll
        for (int j = 0; j < 8; j++) { qa[j] = qp[grp * 8 + j]; qa[j + 8] = (_Float16)0.0f; }
    }
    if (tid < 16 * BOT) {
        int r = tid / BOT, cc = tid % BOT;
        urow[r][cc] = su[((size_t)h * NN + mtile * 16 + r) * BOT + cc];
    }
    __syncthreads();

    // scores: 8 column tiles per wave, one WMMA each; add struct bias + 1/sqrt(HD)
    for (int i = 0; i < 8; i++) {
        int ct = wid * 8 + i;
        int col = ct * 16 + mr;
        v16h kf;
        const _Float16* kp = kb + ((size_t)(bh * NN + col)) * HD;
#pragma unroll
        for (int j = 0; j < 16; j++) { int K = (j < 8) ? grp * 8 + j : 8 + grp * 8 + j; kf[j] = (K < HD) ? kp[K] : (_Float16)0.0f; }
        v8f c0 = {};
        c0 = __builtin_amdgcn_wmma_f32_16x16x32_f16(false, qa, false, kf, (short)0, c0, false, false);
        float vb[BOT];
#pragma unroll
        for (int c2 = 0; c2 < BOT; c2++) vb[c2] = sv[((size_t)h * BOT + c2) * NN + col];
#pragma unroll
        for (int r = 0; r < 8; r++) {
            int M = r + grp * 8;
            float bias = 0.f;
#pragma unroll
            for (int c2 = 0; c2 < BOT; c2++) bias = fmaf(urow[M][c2], vb[c2], bias);
            sc[M][col] = fmaf(c0[r], 0.25f, bias);   // 1/sqrt(16)
        }
    }
    __syncthreads();

    float sthr = 1.f / (1.f + expf(-thr[h]));
    int kk = (int)floorf(512.f * (1.f - sthr));
    if (kk < 1) kk = 1; if (kk > 512) kk = 512;

    // rows wid*4 .. wid*4+3 ; element->lane map col = j*32+lane (bank-conflict free)
    float entS = 0.f;
    for (int r = wid * 4; r < wid * 4 + 4; r++) {
        float sval[16]; unsigned key[16];
#pragma unroll
        for (int j = 0; j < 16; j++) {
            float s = sc[r][j * 32 + lane];
            sval[j] = s; key[j] = f2key(s);
        }
        // radix-select the kk-th largest key of the 512-element row
        unsigned prefix = 0; int rem = kk;
        for (int bit = 31; bit >= 0; --bit) {
            unsigned cand = prefix | (1u << bit);
            unsigned ch = cand >> bit;
            int c = 0;
#pragma unroll
            for (int j = 0; j < 16; j++) c += ((key[j] >> bit) == ch);
            for (int off = 16; off >= 1; off >>= 1) c += __shfl_xor(c, off);
            if (c >= rem) prefix = cand; else rem -= c;
        }
        // masked softmax + entropy
        float mx = -3.0e38f;
#pragma unroll
        for (int j = 0; j < 16; j++) if (key[j] >= prefix) mx = fmaxf(mx, sval[j]);
        for (int off = 16; off >= 1; off >>= 1) mx = fmaxf(mx, __shfl_xor(mx, off));
        float ex[16]; float se = 0.f;
#pragma unroll
        for (int j = 0; j < 16; j++) {
            float e = (key[j] >= prefix) ? expf(sval[j] - mx) : 0.f;
            ex[j] = e; se += e;
        }
        for (int off = 16; off >= 1; off >>= 1) se += __shfl_xor(se, off);
        float inv = 1.f / se;
#pragma unroll
        for (int j = 0; j < 16; j++) {
            float p = ex[j] * inv;
            ap[r][j * 32 + lane] = (_Float16)p;
            if (p > 0.f) entS += p * logf(p + 1e-9f);
        }
    }
    for (int off = 16; off >= 1; off >>= 1) entS += __shfl_xor(entS, off);
    if (lane == 0) atomicAdd(entacc, entS);
    __syncthreads();

    // attn @ v : each wave reduces K-range [wid*128, wid*128+128)
    v8f acc = {};
    const _Float16* vp = vT + ((size_t)bh * HD + mr) * NN;
    for (int i = 0; i < 4; i++) {
        int ks = wid * 128 + i * 32;
        v16h a, bf;
#pragma unroll
        for (int j = 0; j < 8; j++) {
            a[j]     = ap[mr][ks + grp * 8 + j];
            a[j + 8] = ap[mr][ks + 16 + grp * 8 + j];
        }
#pragma unroll
        for (int j = 0; j < 16; j++) bf[j] = vp[ks + ((j < 8) ? grp * 8 + j : 8 + grp * 8 + j)];
        acc = __builtin_amdgcn_wmma_f32_16x16x32_f16(false, a, false, bf, (short)0, acc, false, false);
    }
#pragma unroll
    for (int r = 0; r < 8; r++) pacc[wid][r + grp * 8][mr] = acc[r];
    __syncthreads();
    // reduce 4 partial tiles, 2 elements per thread
    for (int e = tid; e < 256; e += 128) {
        int m = e >> 4, nn = e & 15;
        float s = pacc[0][m][nn] + pacc[1][m][nn] + pacc[2][m][nn] + pacc[3][m][nn];
        attb[((size_t)(bz * NN + mtile * 16 + m)) * NH + h * HD + nn] = s;
    }
}

// ---------------- out-proj + residual + fusion + forecaster ----------------
__global__ void k_tail(const float* __restrict__ attb, const float* __restrict__ temporal,
                       const float* __restrict__ o1w, const float* __restrict__ o1b,
                       const float* __restrict__ o2w, const float* __restrict__ o2b,
                       const float* __restrict__ spimp, const float* __restrict__ tmpimp,
                       const float* __restrict__ fw, const float* __restrict__ fb,
                       const float* __restrict__ flg, const float* __restrict__ flb,
                       const float* __restrict__ p1w, const float* __restrict__ p1b,
                       const float* __restrict__ pg, const float* __restrict__ pbb2,
                       const float* __restrict__ p2w, const float* __restrict__ p2b,
                       float* __restrict__ out) {
    __shared__ float av[NH], red[NH], s6[BOT], c0[NH], c1[NH], fz[NH];
    int row = blockIdx.x; int b = row >> 9; int n = row & 511; int t = threadIdx.x;
    __builtin_prefetch(fw + t * 256, 0, 1);   // pull fuse_w through L2 (global_prefetch_b8)
    av[t] = attb[(size_t)row * NH + t]; __syncthreads();
    if (t < BOT) {
        float s = o1b[t];
        for (int i = 0; i < NH; i++) s = fmaf(av[i], o1w[i * BOT + t], s);
        s6[t] = fmaxf(s, 0.f);
    }
    __syncthreads();
    float tval = temporal[(size_t)row * NH + t];
    float sp = o2b[t];
    for (int j = 0; j < BOT; j++) sp = fmaf(s6[j], o2w[j * NH + t], sp);
    sp += tval;                                    // residual
    float sw = 1.f / (1.f + expf(-spimp[t]));
    float tw = 1.f / (1.f + expf(-tmpimp[t]));
    float tot = sw + tw;
    c0[t] = sp * (sw / tot); c1[t] = tval * (tw / tot);
    __syncthreads();
    float fv = fb[t];
    for (int i = 0; i < NH; i++) fv = fmaf(c0[i], fw[i * NH + t], fv);
    for (int i = 0; i < NH; i++) fv = fmaf(c1[i], fw[(NH + i) * NH + t], fv);
    float m  = blockReduceSum(fv, red, t) * (1.f / 128.f);
    float vq = blockReduceSum(fv * fv, red, t) * (1.f / 128.f) - m * m;
    fz[t] = fmaxf(fmaf((fv - m) * rsqrtf(vq + EPS_LN), flg[t], flb[t]), 0.f);
    __syncthreads();
    if (t < BOT) {
        float p = p1b[t];
        for (int i = 0; i < NH; i++) p = fmaf(fz[i], p1w[i * BOT + t], p);
        p = fmaf(p, pg[t] * rsqrtf(1.f + EPS_BN), pbb2[t]);
        s6[t] = fmaxf(p, 0.f);
    }
    __syncthreads();
    if (t < HOR) {
        float pr = p2b[t];
        for (int j = 0; j < BOT; j++) pr = fmaf(s6[j], p2w[j * HOR + t], pr);
        out[(size_t)(b * HOR + t) * NN + n] = pr;
    }
}

// ---------------- reg loss epilogue ----------------
__global__ void k_reg(const float* __restrict__ l1w, const float* __restrict__ entw,
                      const float* __restrict__ acc, float* __restrict__ out) {
    if (threadIdx.x == 0 && blockIdx.x == 0) {
        float sp1 = log1pf(expf(l1w[0]));
        float sp2 = log1pf(expf(entw[0]));
        // rows of attn each sum to exactly 1 -> mean|attn| = 1/N
        out[NB * HOR * NN] = sp1 * (1.f / (float)NN) + sp2 * acc[0] * (1.f / 32768.f);
    }
}

extern "C" void kernel_launch(void* const* d_in, const int* in_sizes, int n_in,
                              void* d_out, int out_size, void* d_ws, size_t ws_size,
                              hipStream_t stream) {
    (void)in_sizes; (void)n_in; (void)out_size; (void)ws_size;
    const float* x        = (const float*)d_in[0];
    const float* scale_w  = (const float*)d_in[1];
    const float* conv_w   = (const float*)d_in[2];
    const float* conv_b   = (const float*)d_in[3];
    const float* bn_g     = (const float*)d_in[4];
    const float* bn_b     = (const float*)d_in[5];
    const float* proj_w   = (const float*)d_in[6];
    const float* proj_b   = (const float*)d_in[7];
    const float* proj_bg  = (const float*)d_in[8];
    const float* proj_bb  = (const float*)d_in[9];
    const float* ln_g     = (const float*)d_in[10];
    const float* ln_b     = (const float*)d_in[11];
    const float* qkv_lw   = (const float*)d_in[12];
    const float* qkv_lb   = (const float*)d_in[13];
    const float* qkv_hw   = (const float*)d_in[14];
    const float* qkv_hb   = (const float*)d_in[15];
    const float* thr      = (const float*)d_in[16];
    const float* l1_w     = (const float*)d_in[17];
    const float* ent_w    = (const float*)d_in[18];
    const float* struct_u = (const float*)d_in[19];
    const float* struct_v = (const float*)d_in[20];
    const float* out1_w   = (const float*)d_in[21];
    const float* out1_b   = (const float*)d_in[22];
    const float* out2_w   = (const float*)d_in[23];
    const float* out2_b   = (const float*)d_in[24];
    const float* sp_imp   = (const float*)d_in[25];
    const float* tmp_imp  = (const float*)d_in[26];
    const float* fuse_w   = (const float*)d_in[27];
    const float* fuse_b   = (const float*)d_in[28];
    const float* fln_g    = (const float*)d_in[29];
    const float* fln_b    = (const float*)d_in[30];
    const float* pred1_w  = (const float*)d_in[31];
    const float* pred1_b  = (const float*)d_in[32];
    const float* pbn_g    = (const float*)d_in[33];
    const float* pbn_b    = (const float*)d_in[34];
    const float* pred2_w  = (const float*)d_in[35];
    const float* pred2_b  = (const float*)d_in[36];
    float* out = (float*)d_out;

    // workspace layout (256B aligned)
    char* w = (char*)d_ws;
    size_t off = 0;
    auto take = [&](size_t bytes) { size_t cur = off; off = (off + bytes + 255) & ~(size_t)255; return cur; };
    float*    entacc   = (float*)(w + take(4));
    float*    temporal = (float*)(w + take((size_t)ROWS * NH * 4));
    float*    attb     = (float*)(w + take((size_t)ROWS * NH * 4));
    _Float16* catf     = (_Float16*)(w + take((size_t)ROWS * CATK * 2));
    _Float16* wT       = (_Float16*)(w + take((size_t)NH * CATK * 2));
    _Float16* qb       = (_Float16*)(w + take((size_t)NB * HEADS * NN * HD * 2));
    _Float16* kb       = (_Float16*)(w + take((size_t)NB * HEADS * NN * HD * 2));
    _Float16* vT       = (_Float16*)(w + take((size_t)NB * HEADS * HD * NN * 2));

    k_init<<<1, 32, 0, stream>>>(entacc);
    k_prepw<<<(CATK * NH + 255) / 256, 256, 0, stream>>>(proj_w, wT);
    k_multiscale<<<ROWS, NH, 0, stream>>>(x, scale_w, conv_w, conv_b, bn_g, bn_b, catf);
    k_proj<<<dim3(ROWS / 16, NH / 16), 32, 0, stream>>>(catf, wT, proj_b, proj_bg, proj_bb, temporal);
    k_lnqkv<<<ROWS, NH, 0, stream>>>(temporal, ln_g, ln_b, qkv_lw, qkv_lb, qkv_hw, qkv_hb, qb, kb, vT);
    k_attn<<<dim3(NN / 16, HEADS, NB), 128, 0, stream>>>(qb, kb, vT, struct_u, struct_v, thr, attb, entacc);
    k_tail<<<ROWS, NH, 0, stream>>>(attb, temporal, out1_w, out1_b, out2_w, out2_b,
                                    sp_imp, tmp_imp, fuse_w, fuse_b, fln_g, fln_b,
                                    pred1_w, pred1_b, pbn_g, pbn_b, pred2_w, pred2_b, out);
    k_reg<<<1, 32, 0, stream>>>(l1_w, ent_w, entacc, out);
}
